// DenseSAKELayer_75531294868027
// MI455X (gfx1250) — matrix operand order repaired and checked by reading, not compiled, not run on hardware
//
#include <hip/hip_runtime.h>
#include <hip/hip_bf16.h>

#define NDIM 512
#define BDIM 2
#define FDIM 64
#define HDIM 64
#define CDIM 32
#define INFV 1.0e5f
#define EPSV 1.0e-5f

typedef __bf16 bf16_t;
typedef bf16_t bf16x8  __attribute__((ext_vector_type(8)));
typedef bf16_t bf16x16 __attribute__((ext_vector_type(16)));
typedef float  f32x8   __attribute__((ext_vector_type(8)));

struct SmemT {
    bf16_t   he[NDIM * HDIM];          // 64 KB : h_e for all j (bf16)
    unsigned packWb[2048];             // ew1[64:128] packed B-fragments
    unsigned packE2[2048];             // ew2
    unsigned packC1[2048];             // cw1
    unsigned packK1[2048];             // kw1
    unsigned packK2[1024];             // kw2 (64x32)
    bf16_t   stag[8][16 * HDIM];       // per-wave staging (16 KB)
    float    darr[NDIM][3];
    float    narr[NDIM];
    float    marr[NDIM];
    float    semv[NDIM];
    float    att[NDIM];
    float    const_i[HDIM];
    float    wnorm[HDIM];
    float    eb2v[HDIM], cb1v[HDIM], cw2v[HDIM], kb1v[HDIM], swv[HDIM];
    float    kb2v[CDIM];
    float    comb[CDIM][3];
    float    xacc[4];
    float    red[256];
    float    agg4[4][HDIM];
    float    hagg[HDIM];
    float    nin[2 * HDIM + FDIM];
    float    t1[HDIM];
};

__device__ inline float frcp(float x) { return __builtin_amdgcn_rcpf(x); }
__device__ inline float siluf(float x) { return x * frcp(1.0f + __expf(-x)); }

__device__ inline bf16_t to_bf(float x) { return (bf16_t)x; }   // native v_cvt to bf16 (RNE)
__device__ inline float bf_to_f(bf16_t x) { return (float)x; }

// per-wave LDS fence: wait own DS ops, block compiler motion
__device__ inline void wave_lds_fence() {
    __builtin_amdgcn_wave_barrier();
    asm volatile("s_wait_dscnt 0" ::: "memory");
    __builtin_amdgcn_wave_barrier();
}

__device__ inline f32x8 wmma_bf16(bf16x16 a, bf16x16 b, f32x8 c) {
    return __builtin_amdgcn_wmma_f32_16x16x32_bf16(false, a, false, b, (short)0, c, false, false);
}

// pack fp32 weight [rows>=row_off+64][Ncols] into B-fragment order:
// dword idx = ((kb*NB+nb)*32 + lane)*8 + v ; pair (k, k+1), k = kb*32 + 8*(lane>>4) + 2*(v&3) + 16*(v>>2)
__device__ inline void pack_weight(unsigned* dst, const float* src, int Ncols, int NB, int row_off) {
    int total = 2 * NB * 32 * 8;
    for (int idx = threadIdx.x; idx < total; idx += 256) {
        int v  = idx & 7;
        int ln = (idx >> 3) & 31;
        int fi = idx >> 8;
        int nb = fi % NB, kb = fi / NB;
        int k  = kb * 32 + ((ln >> 4) << 3) + 2 * (v & 3) + ((v >> 2) << 4);
        int c  = nb * 16 + (ln & 15);
        unsigned short ua = __builtin_bit_cast(unsigned short, to_bf(src[(row_off + k) * Ncols + c]));
        unsigned short ub = __builtin_bit_cast(unsigned short, to_bf(src[(row_off + k + 1) * Ncols + c]));
        dst[idx] = (unsigned)ua | ((unsigned)ub << 16);
    }
}

__device__ inline bf16x16 load_b_frag(const unsigned* pack, int kb, int nb, int NB, int lane) {
    const unsigned* p = pack + ((((kb * NB + nb) * 32) + lane) << 3);
    return *(const bf16x16*)p;
}

// A fragment from bf16 row-major [16][64] (row stride 64)
__device__ inline bf16x16 load_a_frag_lds(const bf16_t* base, int kb, int lane) {
    const bf16_t* p = base + (lane & 15) * HDIM + kb * 32 + ((lane >> 4) << 3);
    bf16x8 lo = *(const bf16x8*)p;
    bf16x8 hi = *(const bf16x8*)(p + 16);
    bf16x16 r;
#pragma unroll
    for (int t = 0; t < 8; ++t) { r[t] = lo[t]; r[8 + t] = hi[t]; }
    return r;
}

// A fragment straight from global fp32 rows (row stride 64), converted to bf16
__device__ inline bf16x16 load_a_frag_gl(const float* base, int kb, int lane) {
    const float* p = base + (lane & 15) * FDIM + kb * 32 + ((lane >> 4) << 3);
    bf16x16 r;
#pragma unroll
    for (int t = 0; t < 8; ++t) { r[t] = to_bf(p[t]); r[8 + t] = to_bf(p[16 + t]); }
    return r;
}

__device__ inline float block_reduce(SmemT* s, float v, bool ismax) {
    int tid = threadIdx.x;
    s->red[tid] = v;
    __syncthreads();
    for (int st = 128; st > 0; st >>= 1) {
        if (tid < st) {
            float a = s->red[tid], b = s->red[tid + st];
            s->red[tid] = ismax ? fmaxf(a, b) : (a + b);
        }
        __syncthreads();
    }
    float r = s->red[0];
    __syncthreads();
    return r;
}

__global__ __launch_bounds__(256)
void sake_fused_kernel(const float* __restrict__ h, const float* __restrict__ x,
                       const float* __restrict__ mask,
                       const float* __restrict__ ew1, const float* __restrict__ eb1,
                       const float* __restrict__ ew2, const float* __restrict__ eb2,
                       const float* __restrict__ cw1, const float* __restrict__ cb1,
                       const float* __restrict__ cw2,
                       const float* __restrict__ kw1, const float* __restrict__ kb1,
                       const float* __restrict__ kw2, const float* __restrict__ kb2,
                       const float* __restrict__ pw1, const float* __restrict__ pb1,
                       const float* __restrict__ pw2, const float* __restrict__ pb2,
                       const float* __restrict__ sw, const float* __restrict__ sb,
                       const float* __restrict__ nw1, const float* __restrict__ nb1,
                       const float* __restrict__ nw2, const float* __restrict__ nb2,
                       const float* __restrict__ log_gamma,
                       float* __restrict__ hout, float* __restrict__ xout) {
    __shared__ SmemT sm;
    const int i    = blockIdx.x;
    const int b    = blockIdx.y;
    const int tid  = threadIdx.x;
    const int lane = tid & 31;
    const int wv   = tid >> 5;

    const float* hb     = h + (size_t)b * NDIM * FDIM;
    const float* xb     = x + (size_t)b * NDIM * 3;
    const float* hi_row = hb + i * FDIM;
    const float xi0 = xb[i * 3 + 0], xi1 = xb[i * 3 + 1], xi2 = xb[i * 3 + 2];

    // ---- setup: geometry, mask, constants, packed weights ----
    for (int j = tid; j < NDIM; j += 256) {
        float d0 = xb[j * 3 + 0] - xi0, d1 = xb[j * 3 + 1] - xi1, d2 = xb[j * 3 + 2] - xi2;
        sm.darr[j][0] = d0; sm.darr[j][1] = d1; sm.darr[j][2] = d2;
        sm.narr[j] = sqrtf(d0 * d0 + d1 * d1 + d2 * d2 + EPSV);
        sm.marr[j] = mask[((size_t)b * NDIM + i) * NDIM + j];
    }
    if (tid < HDIM) {
        int n = tid;
        float s = eb1[n];
        for (int k = 0; k < FDIM; ++k) s += hi_row[k] * ew1[k * HDIM + n];
        sm.const_i[n] = s;
        sm.wnorm[n]   = ew1[128 * HDIM + n];
        sm.eb2v[n] = eb2[n]; sm.cb1v[n] = cb1[n]; sm.cw2v[n] = cw2[n];
        sm.kb1v[n] = kb1[n]; sm.swv[n]  = sw[n];
        if (n < CDIM) sm.kb2v[n] = kb2[n];
    }
    if (tid < CDIM * 3) ((float*)sm.comb)[tid] = 0.0f;
    if (tid < 4) sm.xacc[tid] = 0.0f;
    pack_weight(sm.packWb, ew1, HDIM, 4, 64);   // h_j half of ew1
    pack_weight(sm.packE2, ew2, HDIM, 4, 0);
    pack_weight(sm.packC1, cw1, HDIM, 4, 0);
    pack_weight(sm.packK1, kw1, HDIM, 4, 0);
    pack_weight(sm.packK2, kw2, CDIM, 2, 0);
    __syncthreads();

    // ---- pass 1: fused edge pipeline, one wave per 16-j tile ----
    float cacc[2][3] = {};
    float txacc[3]   = {};
    bf16_t* stg = sm.stag[wv];
    const int rbase = (lane >> 4) << 3;   // 0 or 8
    const int nlo   = lane & 15;

    for (int tile = wv; tile < NDIM / 16; tile += 8) {
        const int j0 = tile * 16;
        // GEMM1: z1 = silu(h_j @ Wb + norm*wn + const_i)
        bf16x16 a0 = load_a_frag_gl(hb + j0 * FDIM, 0, lane);
        bf16x16 a1 = load_a_frag_gl(hb + j0 * FDIM, 1, lane);
#pragma unroll
        for (int nb = 0; nb < 4; ++nb) {
            f32x8 acc = {};
            acc = wmma_bf16(a0, load_b_frag(sm.packWb, 0, nb, 4, lane), acc);
            acc = wmma_bf16(a1, load_b_frag(sm.packWb, 1, nb, 4, lane), acc);
            int n = nb * 16 + nlo;
            float ci = sm.const_i[n], wn = sm.wnorm[n];
#pragma unroll
            for (int r = 0; r < 8; ++r) {
                int j = j0 + r + rbase;
                stg[(r + rbase) * HDIM + n] = to_bf(siluf(acc[r] + ci + sm.narr[j] * wn));
            }
        }
        wave_lds_fence();
        // GEMM2: h_e = silu(z1 @ ew2 + eb2); sem partial dot with sw
        bf16x16 za0 = load_a_frag_lds(stg, 0, lane);
        bf16x16 za1 = load_a_frag_lds(stg, 1, lane);
        float psem[8] = {};
#pragma unroll
        for (int nb = 0; nb < 4; ++nb) {
            f32x8 acc = {};
            acc = wmma_bf16(za0, load_b_frag(sm.packE2, 0, nb, 4, lane), acc);
            acc = wmma_bf16(za1, load_b_frag(sm.packE2, 1, nb, 4, lane), acc);
            int n = nb * 16 + nlo;
            float bb = sm.eb2v[n], swn = sm.swv[n];
#pragma unroll
            for (int r = 0; r < 8; ++r) {
                float hev = siluf(acc[r] + bb);
                sm.he[(j0 + r + rbase) * HDIM + n] = to_bf(hev);
                psem[r] += hev * swn;
            }
        }
#pragma unroll
        for (int r = 0; r < 8; ++r) {
            float v = psem[r];
            v += __shfl_xor(v, 1); v += __shfl_xor(v, 2);
            v += __shfl_xor(v, 4); v += __shfl_xor(v, 8);
            if (nlo == 0) {
                float s = v + sb[0];
                sm.semv[j0 + r + rbase] = (s >= 0.0f) ? s : 0.01f * s;
            }
        }
        wave_lds_fence();
        // shared A fragments of h_e for GEMM3 and GEMM4
        bf16x16 ha0 = load_a_frag_lds(&sm.he[j0 * HDIM], 0, lane);
        bf16x16 ha1 = load_a_frag_lds(&sm.he[j0 * HDIM], 1, lane);
        // GEMM3: coordinate scalar t = (silu(h_e@cw1+cb1)) @ cw2
        float pt[8] = {};
#pragma unroll
        for (int nb = 0; nb < 4; ++nb) {
            f32x8 acc = {};
            acc = wmma_bf16(ha0, load_b_frag(sm.packC1, 0, nb, 4, lane), acc);
            acc = wmma_bf16(ha1, load_b_frag(sm.packC1, 1, nb, 4, lane), acc);
            int n = nb * 16 + nlo;
            float cb = sm.cb1v[n], c2 = sm.cw2v[n];
#pragma unroll
            for (int r = 0; r < 8; ++r) pt[r] += siluf(acc[r] + cb) * c2;
        }
#pragma unroll
        for (int r = 0; r < 8; ++r) {
            float v = pt[r];
            v += __shfl_xor(v, 1); v += __shfl_xor(v, 2);
            v += __shfl_xor(v, 4); v += __shfl_xor(v, 8);
            if (nlo == 0) {
                int j = j0 + r + rbase;
                txacc[0] += sm.darr[j][0] * v;
                txacc[1] += sm.darr[j][1] * v;
                txacc[2] += sm.darr[j][2] * v;
            }
        }
        // GEMM4: v4 = silu(h_e@kw1+kb1) -> staging
#pragma unroll
        for (int nb = 0; nb < 4; ++nb) {
            f32x8 acc = {};
            acc = wmma_bf16(ha0, load_b_frag(sm.packK1, 0, nb, 4, lane), acc);
            acc = wmma_bf16(ha1, load_b_frag(sm.packK1, 1, nb, 4, lane), acc);
            int n = nb * 16 + nlo;
            float kb = sm.kb1v[n];
#pragma unroll
            for (int r = 0; r < 8; ++r)
                stg[(r + rbase) * HDIM + n] = to_bf(siluf(acc[r] + kb));
        }
        wave_lds_fence();
        // coeff = v4 @ kw2 + kb2 ; accumulate outer product with dirv
        bf16x16 va0 = load_a_frag_lds(stg, 0, lane);
        bf16x16 va1 = load_a_frag_lds(stg, 1, lane);
#pragma unroll
        for (int nb = 0; nb < 2; ++nb) {
            f32x8 acc = {};
            acc = wmma_bf16(va0, load_b_frag(sm.packK2, 0, nb, 2, lane), acc);
            acc = wmma_bf16(va1, load_b_frag(sm.packK2, 1, nb, 2, lane), acc);
            float kb2n = sm.kb2v[nb * 16 + nlo];
#pragma unroll
            for (int r = 0; r < 8; ++r) {
                int j = j0 + r + rbase;
                float nrm = sm.narr[j];
                float w = (acc[r] + kb2n) * sm.marr[j] * frcp(nrm * nrm + EPSV);
                cacc[nb][0] += w * sm.darr[j][0];
                cacc[nb][1] += w * sm.darr[j][1];
                cacc[nb][2] += w * sm.darr[j][2];
            }
        }
    }

    // flush per-lane accumulators (ds_add_f32)
#pragma unroll
    for (int nb = 0; nb < 2; ++nb) {
        atomicAdd(&sm.comb[nb * 16 + nlo][0], cacc[nb][0]);
        atomicAdd(&sm.comb[nb * 16 + nlo][1], cacc[nb][1]);
        atomicAdd(&sm.comb[nb * 16 + nlo][2], cacc[nb][2]);
    }
    if (nlo == 0) {
        atomicAdd(&sm.xacc[0], txacc[0]);
        atomicAdd(&sm.xacc[1], txacc[1]);
        atomicAdd(&sm.xacc[2], txacc[2]);
    }
    __syncthreads();

    // ---- pass 2: three softmaxes over j, aggregation, tail MLPs ----
    const float g = __expf(log_gamma[0]);
    const int jA = tid, jB = tid + 256;
    float e1a = -(sm.narr[jA] + (jA == i ? INFV : 0.0f)) * g;
    float e1b = -(sm.narr[jB] + (jB == i ? INFV : 0.0f)) * g;
    float m1 = block_reduce(&sm, fmaxf(e1a, e1b), true);
    float x1a = __expf(e1a - m1), x1b = __expf(e1b - m1);
    float r1 = frcp(block_reduce(&sm, x1a + x1b, false));
    float p1a = x1a * r1, p1b = x1b * r1;

    float e2a = sm.semv[jA] - (jA == i ? INFV : 0.0f);
    float e2b = sm.semv[jB] - (jB == i ? INFV : 0.0f);
    float m2 = block_reduce(&sm, fmaxf(e2a, e2b), true);
    float x2a = __expf(e2a - m2), x2b = __expf(e2b - m2);
    float r2 = frcp(block_reduce(&sm, x2a + x2b, false));
    float qa = p1a * x2a * r2, qb = p1b * x2b * r2;

    float m3 = block_reduce(&sm, fmaxf(qa, qb), true);
    float x3a = __expf(qa - m3), x3b = __expf(qb - m3);
    float r3 = frcp(block_reduce(&sm, x3a + x3b, false));
    sm.att[jA] = x3a * r3;
    sm.att[jB] = x3b * r3;
    __syncthreads();

    // h_e_agg = sum_j h_e * att * mask
    {
        int c = tid & 63, ch = tid >> 6;
        float s = 0.0f;
#pragma unroll 4
        for (int j = ch * 128; j < ch * 128 + 128; ++j)
            s += bf_to_f(sm.he[j * HDIM + c]) * sm.att[j] * sm.marr[j];
        sm.agg4[ch][c] = s;
    }
    __syncthreads();
    if (tid < HDIM)
        sm.hagg[tid] = sm.agg4[0][tid] + sm.agg4[1][tid] + sm.agg4[2][tid] + sm.agg4[3][tid];
    __syncthreads();

    // post_norm_mlp layer 1 from comb_norm
    if (tid < HDIM) {
        int n = tid;
        float s = pb1[n];
        for (int c = 0; c < CDIM; ++c) {
            float cn = sm.comb[c][0] * sm.comb[c][0] + sm.comb[c][1] * sm.comb[c][1] +
                       sm.comb[c][2] * sm.comb[c][2];
            s += cn * pw1[c * HDIM + n];
        }
        sm.t1[n] = siluf(s);
    }
    __syncthreads();
    if (tid < HDIM) {
        int n = tid;
        float s = pb2[n];
        for (int k = 0; k < HDIM; ++k) s += sm.t1[k] * pw2[k * HDIM + n];
        sm.nin[n]               = hi_row[n];   // h
        sm.nin[FDIM + n]        = sm.hagg[n];  // h_e_agg
        sm.nin[FDIM + HDIM + n] = s;           // h_comb
    }
    __syncthreads();
    if (tid < HDIM) {
        int n = tid;
        float s = nb1[n];
        for (int k = 0; k < 2 * HDIM + FDIM; ++k) s += sm.nin[k] * nw1[k * HDIM + n];
        sm.t1[n] = siluf(s);
    }
    __syncthreads();
    if (tid < FDIM) {
        int f = tid;
        float s = nb2[f];
        for (int k = 0; k < HDIM; ++k) s += sm.t1[k] * nw2[k * FDIM + f];
        hout[((size_t)b * NDIM + i) * FDIM + f] = hi_row[f] + s;
    }
    if (tid < 3)
        xout[((size_t)b * NDIM + i) * 3 + tid] = xb[i * 3 + tid] + sm.xacc[tid] * (1.0f / (float)NDIM);
}

extern "C" void kernel_launch(void* const* d_in, const int* in_sizes, int n_in,
                              void* d_out, int out_size, void* d_ws, size_t ws_size,
                              hipStream_t stream) {
    const float* h   = (const float*)d_in[0];
    const float* x   = (const float*)d_in[1];
    const float* msk = (const float*)d_in[2];
    const float* ew1 = (const float*)d_in[3];
    const float* eb1 = (const float*)d_in[4];
    const float* ew2 = (const float*)d_in[5];
    const float* eb2 = (const float*)d_in[6];
    const float* cw1 = (const float*)d_in[7];
    const float* cb1 = (const float*)d_in[8];
    const float* cw2 = (const float*)d_in[9];
    const float* kw1 = (const float*)d_in[10];
    const float* kb1 = (const float*)d_in[11];
    const float* kw2 = (const float*)d_in[12];
    const float* kb2 = (const float*)d_in[13];
    const float* pw1 = (const float*)d_in[14];
    const float* pb1 = (const float*)d_in[15];
    const float* pw2 = (const float*)d_in[16];
    const float* pb2 = (const float*)d_in[17];
    const float* sw  = (const float*)d_in[18];
    const float* sb  = (const float*)d_in[19];
    const float* nw1 = (const float*)d_in[20];
    const float* nb1 = (const float*)d_in[21];
    const float* nw2 = (const float*)d_in[22];
    const float* nb2 = (const float*)d_in[23];
    const float* lg  = (const float*)d_in[24];

    float* hout = (float*)d_out;
    float* xout = hout + (size_t)BDIM * NDIM * FDIM;

    dim3 grid(NDIM, BDIM), block(256);
    sake_fused_kernel<<<grid, block, 0, stream>>>(h, x, msk, ew1, eb1, ew2, eb2, cw1, cb1, cw2,
                                                  kw1, kb1, kw2, kb2, pw1, pb1, pw2, pb2,
                                                  sw, sb, nw1, nb1, nw2, nb2, lg, hout, xout);
}
